// SA_module_16320875725212
// MI455X (gfx1250) — matrix operand order repaired and checked by reading, not compiled, not run on hardware
//
#include <hip/hip_runtime.h>
#include <hip/hip_bf16.h>

// ---------------------------------------------------------------------------
// Problem constants (from the reference)
// ---------------------------------------------------------------------------
#define B_   8
#define P1_  8192
#define P2_  2048
#define KNN_ 32
#define C1_  64
#define M1_  64
#define M2_  64
#define M3_  128
#define BN_EPS_ 1e-5f

// padded Cin for layer1: 64 feat + 3 xyz + 29 zero = 96 (3 k-steps of 32)
#define CIN1P 96

typedef __attribute__((ext_vector_type(16))) _Float16 v16h;
typedef __attribute__((ext_vector_type(8)))  float    v8f;
typedef __attribute__((ext_vector_type(4)))  int      v4i;

// packed weight sizes (halves)
#define W1P_H 6144      // 4 tiles * 3 ksteps * 32 lanes * 16
#define W2P_H 4096      // 4 tiles * 2 ksteps * 512
#define W3P_H 8192      // 8 tiles * 2 ksteps * 512
#define WALL_H (W1P_H + W2P_H + W3P_H)            // 18432 halves = 36 KB

// workspace layout (bytes): packed weights contiguous, then biases
#define WS_W2_OFF  (W1P_H * 2)                    // 12288
#define WS_W3_OFF  ((W1P_H + W2P_H) * 2)          // 20480
#define WS_B_OFF   (WALL_H * 2)                   // 36864

// output layout (elements, mixed dtype reinterpreted)
#define OUT_NN_ELEMS   (B_ * P2_ * KNN_)          // 524288 int32
#define OUT_XYZ_ELEMS  (B_ * P2_ * 3)             // 49152 f32
// new_feature follows: B_*P2_*M3_ f32

static __device__ __forceinline__ void lds_fence() {
    // wait for all outstanding LDS ops from this wave; also a compiler fence
    asm volatile("s_wait_dscnt 0x0" ::: "memory");
}

// ---------------------------------------------------------------------------
// Weight packing: fold BN (inference) into W and b, convert to f16, and lay
// out in this file's per-lane WMMA B layout (contiguous k, as load_a_frag):
//   entry e = ((t*KS + ks)*32 + lane)*16 + h
//   N = t*16 + (lane & 15)
//   K = ks*32 + (lane >> 4)*16 + h
// ---------------------------------------------------------------------------
static __device__ void pack_layer(const float* w, const float* b, const float* g,
                                  const float* be, const float* rm, const float* rv,
                                  int Cout, int Cin, int KS,
                                  _Float16* wp, float* bp, int tid, int nth) {
    const int ntiles = Cout >> 4;
    const int total  = ntiles * KS * 32 * 16;
    for (int e = tid; e < total; e += nth) {
        int h  = e & 15;
        int l  = (e >> 4) & 31;
        int r  = e >> 9;
        int ks = r % KS;
        int t  = r / KS;
        int N  = t * 16 + (l & 15);
        int K  = ks * 32 + ((l >> 4) << 4) + h;
        float s = g[N] * rsqrtf(rv[N] + BN_EPS_);
        float v = (K < Cin) ? (w[N * Cin + K] * s) : 0.0f;
        wp[e] = (_Float16)v;
    }
    for (int o = tid; o < Cout; o += nth) {
        float s = g[o] * rsqrtf(rv[o] + BN_EPS_);
        bp[o] = (b[o] - rm[o]) * s + be[o];
    }
}

__global__ __launch_bounds__(256) void pack_weights_kernel(
    const float* w1, const float* b1, const float* g1, const float* be1, const float* rm1, const float* rv1,
    const float* w2, const float* b2, const float* g2, const float* be2, const float* rm2, const float* rv2,
    const float* w3, const float* b3, const float* g3, const float* be3, const float* rm3, const float* rv3,
    _Float16* w1p, _Float16* w2p, _Float16* w3p, float* biasp) {
    const int tid = threadIdx.x, nth = blockDim.x;
    pack_layer(w1, b1, g1, be1, rm1, rv1, M1_, C1_ + 3, 3, w1p, biasp,           tid, nth);
    pack_layer(w2, b2, g2, be2, rm2, rv2, M2_, M1_,     2, w2p, biasp + M1_,     tid, nth);
    pack_layer(w3, b3, g3, be3, rm3, rv3, M3_, M2_,     2, w3p, biasp + M1_+M2_, tid, nth);
}

// ---------------------------------------------------------------------------
// KNN kernel: 4 waves per block, one wave per query. Candidate xyz staged in
// LDS in chunks; per-lane sorted top-32 lists in LDS ([entry][lane], bank
// conflict free); final wave-wide merge via shfl_xor argmin (sorted output,
// matching top_k(-D) ascending-distance order).
// ---------------------------------------------------------------------------
#define KNN_CH 1024

__global__ __launch_bounds__(128) void knn_kernel(const float* __restrict__ xyz,
                                                  const int*   __restrict__ sidx,
                                                  int*   __restrict__ nn_out,
                                                  float* __restrict__ nxyz_out) {
    __shared__ float chunk[KNN_CH * 3];
    __shared__ float ldist[4][KNN_][32];
    __shared__ int   lidx [4][KNN_][32];

    const int tid  = threadIdx.x;
    const int lane = tid & 31;
    const int wv   = tid >> 5;
    const int b    = blockIdx.x >> 9;                    // 512 blocks per batch
    const int q    = ((blockIdx.x & 511) << 2) + wv;

    const float* xb = xyz + (size_t)b * P1_ * 3;
    const int si = sidx[b * P2_ + q];
    const float qx = xb[si * 3 + 0];
    const float qy = xb[si * 3 + 1];
    const float qz = xb[si * 3 + 2];
    if (lane < 3) nxyz_out[((size_t)b * P2_ + q) * 3 + lane] = xb[si * 3 + lane];

    for (int e = 0; e < KNN_; ++e) { ldist[wv][e][lane] = 3.4e38f; lidx[wv][e][lane] = 0; }

    for (int cb = 0; cb < P1_; cb += KNN_CH) {
        __syncthreads();
        for (int i = tid; i < KNN_CH * 3; i += 128) chunk[i] = xb[cb * 3 + i];
        __syncthreads();
        for (int p = lane; p < KNN_CH; p += 32) {
            const float dx = chunk[p * 3 + 0] - qx;
            const float dy = chunk[p * 3 + 1] - qy;
            const float dz = chunk[p * 3 + 2] - qz;
            const float d  = dx * dx + dy * dy + dz * dz;
            if (d < ldist[wv][KNN_ - 1][lane]) {       // beats current worst?
                int pos = KNN_ - 1;
                while (pos > 0 && ldist[wv][pos - 1][lane] > d) {
                    ldist[wv][pos][lane] = ldist[wv][pos - 1][lane];
                    lidx [wv][pos][lane] = lidx [wv][pos - 1][lane];
                    --pos;
                }
                ldist[wv][pos][lane] = d;
                lidx [wv][pos][lane] = cb + p;
            }
        }
    }
    lds_fence();

    // merge 32 sorted per-lane lists -> global sorted top-32
    int head = 0;
    int result = 0;
    for (int r = 0; r < KNN_; ++r) {
        const float myd = (head < KNN_) ? ldist[wv][head][lane] : 3.4e38f;
        const int   myi = (head < KNN_) ? lidx [wv][head][lane] : 0;
        float bd = myd; int bl = lane;
        #pragma unroll
        for (int off = 16; off > 0; off >>= 1) {
            const float od = __shfl_xor(bd, off);
            const int   ol = __shfl_xor(bl, off);
            if (od < bd || (od == bd && ol < bl)) { bd = od; bl = ol; }
        }
        const int widx = __shfl(myi, bl);
        if (lane == bl) ++head;
        if (lane == r) result = widx;
    }
    nn_out[((size_t)b * P2_ + q) * KNN_ + lane] = result;
}

// ---------------------------------------------------------------------------
// WMMA fragment helpers
// ---------------------------------------------------------------------------
// A-fragment, contiguous k order (same as packed B): lane m = lane&15, K-half =
// lane>>4; VGPR v: K = (lane>>4)*16 + v*2 (+0,+1).
static __device__ __forceinline__ v16h load_a_frag(const _Float16* buf, int ldh,
                                                   int lane, int ks) {
    union { unsigned u[8]; v16h v; } f;
    const int m  = lane & 15;
    const int kh = lane >> 4;
    const _Float16* rp = buf + m * ldh + ks * 32;
    #pragma unroll
    for (int v0 = 0; v0 < 8; ++v0) {
        const int k0 = (kh << 4) + (v0 << 1);
        f.u[v0] = *(const unsigned*)(rp + k0);
    }
    return f.v;
}

// B fragment from the block-shared LDS weight cache: one aligned 32 B chunk
// per lane (2x ds_load_b128).
static __device__ __forceinline__ v16h load_b_frag(const _Float16* wlds, int t, int ks,
                                                   int KS, int lane) {
    return *(const v16h*)(wlds + (((t * KS) + ks) * 32 + lane) * 16);
}

static __device__ __forceinline__ v8f wmma_f16(v16h a, v16h b, v8f c) {
    return __builtin_amdgcn_wmma_f32_16x16x32_f16(false, a, false, b,
                                                  (short)0, c, false, false);
}

// ---------------------------------------------------------------------------
// Fused gather + 3-layer MLP + max-pool. One wave per query (K=32 neighbors
// = two 16-row M tiles). Packed weights (36 KB) staged once per block into
// LDS via the CDNA5 async global->LDS path; activations bounce through
// per-wave LDS buffers.
// ---------------------------------------------------------------------------
__global__ __launch_bounds__(128) void mlp_kernel(const float* __restrict__ xyz,
                                                  const float* __restrict__ feat,
                                                  const int*   __restrict__ nn,
                                                  const float* __restrict__ nxyz,
                                                  float* __restrict__ outfeat,
                                                  const _Float16* __restrict__ wpk,
                                                  const float* __restrict__ biasp) {
    __shared__ alignas(32) _Float16 wlds[WALL_H];       // 36 KB weight cache
    __shared__ alignas(32) _Float16 xa[4][16 * CIN1P];  // 3 KB / wave
    __shared__ alignas(32) _Float16 ya[4][16 * 64];     // 2 KB / wave

    const int tid  = threadIdx.x;
    const int lane = tid & 31;
    const int wv   = tid >> 5;
    const int wid  = blockIdx.x * 4 + wv;
    const int b    = wid >> 11;              // 2048 queries per batch
    const int q    = wid & 2047;
    const int m    = lane & 15;
    const int kh   = lane >> 4;

    // ---- stage packed weights into LDS (async CDNA5 path when available) ---
    {
        const int n128 = (WALL_H * 2) / 16;  // 2304 16-byte chunks
#if __has_builtin(__builtin_amdgcn_global_load_async_to_lds_b128)
        for (int i = tid; i < n128; i += 128) {
            __builtin_amdgcn_global_load_async_to_lds_b128(
                (__attribute__((address_space(1))) v4i*)((const char*)wpk + i * 16),
                (__attribute__((address_space(3))) v4i*)((char*)wlds + i * 16),
                0, 0);
        }
        asm volatile("s_wait_asynccnt 0x0" ::: "memory");
#else
        const uint4* src = (const uint4*)wpk;
        uint4*       dst = (uint4*)wlds;
        for (int i = tid; i < n128; i += 128) dst[i] = src[i];
#endif
    }
    __syncthreads();

    const int*   nnq = nn   + ((size_t)b * P2_ + q) * KNN_;
    const float* nxq = nxyz + ((size_t)b * P2_ + q) * 3;
    const float qx = nxq[0], qy = nxq[1], qz = nxq[2];

    // prefetch this wave's 32 gathered feature rows while weights settle
    {
        const int j = nnq[lane];
        __builtin_prefetch(feat + ((size_t)b * P1_ + j) * C1_, 0, 1);
    }

    const _Float16* w1l = wlds;
    const _Float16* w2l = wlds + W1P_H;
    const _Float16* w3l = wlds + W1P_H + W2P_H;
    const float* bias1 = biasp;
    const float* bias2 = biasp + M1_;
    const float* bias3 = biasp + M1_ + M2_;

    float pooled[8];
    #pragma unroll
    for (int t = 0; t < 8; ++t) pooled[t] = 0.0f;   // ReLU output >= 0

    for (int mt = 0; mt < 2; ++mt) {
        // ---- stage X tile: row m, columns [kh*48, kh*48+48) -----------------
        {
            const int  k = mt * 16 + m;
            const int  j = nnq[k];
            const float* fr = feat + ((size_t)b * P1_ + j) * C1_;
            const float* xr = xyz  + ((size_t)b * P1_ + j) * 3;
            _Float16* row = &xa[wv][m * CIN1P];
            #pragma unroll 8
            for (int c = 0; c < 48; ++c) {
                const int cc = kh * 48 + c;
                float v;
                if (cc < 64)       v = fr[cc];
                else if (cc == 64) v = xr[0] - qx;
                else if (cc == 65) v = xr[1] - qy;
                else if (cc == 66) v = xr[2] - qz;
                else               v = 0.0f;
                row[cc] = (_Float16)v;
            }
        }
        lds_fence();

        // ---- layer 1: X[16x96] * W1[96x64] ---------------------------------
        v16h a1[3];
        #pragma unroll
        for (int ks = 0; ks < 3; ++ks) a1[ks] = load_a_frag(&xa[wv][0], CIN1P, lane, ks);
        #pragma unroll
        for (int t = 0; t < 4; ++t) {
            v8f c = {};
            #pragma unroll
            for (int ks = 0; ks < 3; ++ks)
                c = wmma_f16(a1[ks], load_b_frag(w1l, t, ks, 3, lane), c);
            const float bb = bias1[t * 16 + m];
            #pragma unroll
            for (int v = 0; v < 8; ++v) {
                const float y = fmaxf(c[v] + bb, 0.0f);
                ya[wv][(kh * 8 + v) * 64 + t * 16 + m] = (_Float16)y;
            }
        }
        lds_fence();

        // ---- layer 2: Y1[16x64] * W2[64x64] --------------------------------
        v16h a2[2];
        #pragma unroll
        for (int ks = 0; ks < 2; ++ks) a2[ks] = load_a_frag(&ya[wv][0], 64, lane, ks);
        #pragma unroll
        for (int t = 0; t < 4; ++t) {
            v8f c = {};
            #pragma unroll
            for (int ks = 0; ks < 2; ++ks)
                c = wmma_f16(a2[ks], load_b_frag(w2l, t, ks, 2, lane), c);
            const float bb = bias2[t * 16 + m];
            #pragma unroll
            for (int v = 0; v < 8; ++v) {
                const float y = fmaxf(c[v] + bb, 0.0f);
                xa[wv][(kh * 8 + v) * 64 + t * 16 + m] = (_Float16)y;   // reuse xa
            }
        }
        lds_fence();

        // ---- layer 3: Y2[16x64] * W3[64x128] + max over the 16 rows --------
        v16h a3[2];
        #pragma unroll
        for (int ks = 0; ks < 2; ++ks) a3[ks] = load_a_frag(&xa[wv][0], 64, lane, ks);
        #pragma unroll
        for (int t = 0; t < 8; ++t) {
            v8f c = {};
            #pragma unroll
            for (int ks = 0; ks < 2; ++ks)
                c = wmma_f16(a3[ks], load_b_frag(w3l, t, ks, 2, lane), c);
            const float bb = bias3[t * 16 + m];
            float mx = 0.0f;
            #pragma unroll
            for (int v = 0; v < 8; ++v) mx = fmaxf(mx, fmaxf(c[v] + bb, 0.0f));
            mx = fmaxf(mx, __shfl_xor(mx, 16));      // combine M 0-7 with 8-15
            pooled[t] = fmaxf(pooled[t], mx);
        }
        lds_fence();   // protect xa WAR before next tile's staging
    }

    if (lane < 16) {
        float* op = outfeat + ((size_t)b * P2_ + q) * M3_;
        #pragma unroll
        for (int t = 0; t < 8; ++t) op[t * 16 + lane] = pooled[t];
    }
}

// ---------------------------------------------------------------------------
// Launch
// ---------------------------------------------------------------------------
extern "C" void kernel_launch(void* const* d_in, const int* in_sizes, int n_in,
                              void* d_out, int out_size, void* d_ws, size_t ws_size,
                              hipStream_t stream) {
    (void)in_sizes; (void)n_in; (void)out_size; (void)ws_size;

    const float* xyz  = (const float*)d_in[0];
    const float* feat = (const float*)d_in[1];
    const int*   sidx = (const int*)d_in[2];

    const float* w1 = (const float*)d_in[3];
    const float* b1 = (const float*)d_in[4];
    const float* g1 = (const float*)d_in[5];
    const float* be1= (const float*)d_in[6];
    const float* rm1= (const float*)d_in[7];
    const float* rv1= (const float*)d_in[8];
    const float* w2 = (const float*)d_in[9];
    const float* b2 = (const float*)d_in[10];
    const float* g2 = (const float*)d_in[11];
    const float* be2= (const float*)d_in[12];
    const float* rm2= (const float*)d_in[13];
    const float* rv2= (const float*)d_in[14];
    const float* w3 = (const float*)d_in[15];
    const float* b3 = (const float*)d_in[16];
    const float* g3 = (const float*)d_in[17];
    const float* be3= (const float*)d_in[18];
    const float* rm3= (const float*)d_in[19];
    const float* rv3= (const float*)d_in[20];

    // outputs, flat in return order: nn_idx(int), new_xyz(f32), new_feature(f32)
    int*   nn_out   = (int*)d_out;
    float* nxyz_out = (float*)d_out + OUT_NN_ELEMS;
    float* feat_out = (float*)d_out + OUT_NN_ELEMS + OUT_XYZ_ELEMS;

    _Float16* w1p   = (_Float16*)d_ws;
    _Float16* w2p   = (_Float16*)((char*)d_ws + WS_W2_OFF);
    _Float16* w3p   = (_Float16*)((char*)d_ws + WS_W3_OFF);
    float*    biasp = (float*)   ((char*)d_ws + WS_B_OFF);

    pack_weights_kernel<<<1, 256, 0, stream>>>(
        w1, b1, g1, be1, rm1, rv1,
        w2, b2, g2, be2, rm2, rv2,
        w3, b3, g3, be3, rm3, rv3,
        w1p, w2p, w3p, biasp);

    knn_kernel<<<(B_ * P2_) / 4, 128, 0, stream>>>(xyz, sidx, nn_out, nxyz_out);

    mlp_kernel<<<(B_ * P2_) / 8 * 2, 128, 0, stream>>>(xyz, feat, nn_out, nxyz_out,
                                                       feat_out, (const _Float16*)d_ws,
                                                       biasp);
}
